// NTSLoss_64080912056667
// MI455X (gfx1250) — compile-verified, matrix operand order checked by reading, not run on hardware
//
#include <hip/hip_runtime.h>
#include <hip/hip_bf16.h>

typedef __attribute__((ext_vector_type(16))) _Float16 v16h;
typedef __attribute__((ext_vector_type(8)))  float    v8f;

#define BLK   256
#define C_DIM 10000
#define C4    (C_DIM / 4)   // 2500 float4 per row
#define NITER 10            // ceil(2500 / 256)

// One block per row. Computes row max, sum(exp(x-max)), sum(x), x[target],
// emits smoothed-CE per row (and plain NLL for the 1536 part rows).
__global__ __launch_bounds__(BLK) void nts_row_stats(
    const float* __restrict__ raw, const float* __restrict__ concat,
    const float* __restrict__ part, const int* __restrict__ targets,
    float* __restrict__ ce_all, float* __restrict__ part_nll)
{
    const int row = blockIdx.x;
    const int tid = threadIdx.x;

    const float* src;
    int  tgt;
    bool is_part = false;
    if (row < 256) {
        src = raw + (size_t)row * C_DIM;            tgt = targets[row];
    } else if (row < 512) {
        src = concat + (size_t)(row - 256) * C_DIM; tgt = targets[row - 256];
    } else {
        int r = row - 512;
        src = part + (size_t)r * C_DIM;             tgt = targets[r / 6];
        is_part = true;
    }

    // Pass 1: load row (b128), cache in registers, reduce max.
    const float4* p4 = (const float4*)src;
    float4 vals[NITER];
    float  m = -3.402823466e38f;
#pragma unroll
    for (int k = 0; k < NITER; ++k) {
        int i = tid + k * BLK;
        if (i < C4) {
            float4 v = p4[i];
            vals[k] = v;
            m = fmaxf(m, fmaxf(fmaxf(v.x, v.y), fmaxf(v.z, v.w)));
        }
    }

    __shared__ float s0[BLK];
    __shared__ float s1[BLK];
    s0[tid] = m;
    __syncthreads();
    for (int st = BLK / 2; st > 0; st >>= 1) {
        if (tid < st) s0[tid] = fmaxf(s0[tid], s0[tid + st]);
        __syncthreads();
    }
    const float M = s0[0];
    __syncthreads();

    // Pass 2: from registers (no HBM re-read): sum exp(x-M) and sum x.
    float s = 0.f, t = 0.f;
#pragma unroll
    for (int k = 0; k < NITER; ++k) {
        int i = tid + k * BLK;
        if (i < C4) {
            float4 v = vals[k];
            s += expf(v.x - M) + expf(v.y - M) + expf(v.z - M) + expf(v.w - M);
            t += v.x + v.y + v.z + v.w;
        }
    }
    s0[tid] = s;
    s1[tid] = t;
    __syncthreads();
    for (int st = BLK / 2; st > 0; st >>= 1) {
        if (tid < st) { s0[tid] += s0[tid + st]; s1[tid] += s1[tid + st]; }
        __syncthreads();
    }

    if (tid == 0) {
        float lse    = M + logf(s0[0]);
        float xt     = src[tgt];
        float nll    = lse - xt;
        float smooth = lse - s1[0] * (1.0f / C_DIM);
        ce_all[row]  = 0.9f * nll + 0.1f * smooth;   // label smoothing eps=0.1
        if (is_part) part_nll[row - 512] = nll;
    }
}

// Single block. Per-sample rank hinge loss + WMMA-based global sum.
// With B = ones, D[m][n] = sum_k A[m][k], so sum(all D) = 16 * sum(all A
// slots) regardless of the hardware's A/C lane layouts. Each f32 value is
// split hi/lo (Dekker) across two f16 A-slots to retain f32 accuracy.
__global__ __launch_bounds__(BLK) void nts_finalize(
    const float* __restrict__ part_nll, const float* __restrict__ top_n_prob,
    const float* __restrict__ ce_all, float* __restrict__ out)
{
    const int tid = threadIdx.x;
    __shared__ float rankLds[BLK];
    __shared__ float red[32];

    // ranking loss for sample b = tid (B=256, P=6)
    float pl[6], pp[6];
#pragma unroll
    for (int k = 0; k < 6; ++k) {
        pl[k] = part_nll[tid * 6 + k];
        pp[k] = top_n_prob[tid * 6 + k];
    }
    float r = 0.f;
#pragma unroll
    for (int i = 0; i < 6; ++i) {
#pragma unroll
        for (int j = 0; j < 6; ++j) {
            if (pl[j] > pl[i]) {                 // mask[b,i,j]
                float h = 1.0f - pp[i] + pp[j];  // hinge
                r += (h > 0.f) ? h : 0.f;
            }
        }
    }
    rankLds[tid] = r * (1.0f / 256.0f);          // rank_loss contribution /B
    __syncthreads();

    if (tid < 32) {  // wave 0, EXEC all ones for WMMA
        v8f  acc = {};
        v16h ones;
#pragma unroll
        for (int q = 0; q < 16; ++q) ones[q] = (_Float16)1.0f;

        // 2048 CE values in chunks of 256 (8 values/lane, hi/lo split)
        for (int base = 0; base < 2048; base += 256) {
            v16h a;
#pragma unroll
            for (int q = 0; q < 8; ++q) {
                int   idx = base + tid * 8 + q;
                float w   = (idx < 512) ? (1.0f / 256.0f) : (1.0f / 1536.0f);
                float v   = ce_all[idx] * w;
                _Float16 hi = (_Float16)v;
                float    lo = v - (float)hi;
                a[2 * q]     = hi;
                a[2 * q + 1] = (_Float16)lo;
            }
            acc = __builtin_amdgcn_wmma_f32_16x16x32_f16(
                false, a, false, ones, (short)0, acc, false, false);
        }
        // 256 rank values, one chunk
        {
            v16h a;
#pragma unroll
            for (int q = 0; q < 8; ++q) {
                float v = rankLds[tid * 8 + q];
                _Float16 hi = (_Float16)v;
                float    lo = v - (float)hi;
                a[2 * q]     = hi;
                a[2 * q + 1] = (_Float16)lo;
            }
            acc = __builtin_amdgcn_wmma_f32_16x16x32_f16(
                false, a, false, ones, (short)0, acc, false, false);
        }
        red[tid] = acc[0] + acc[1] + acc[2] + acc[3] +
                   acc[4] + acc[5] + acc[6] + acc[7];
    }
    __syncthreads();

    if (tid == 0) {
        float sTot = 0.f;
#pragma unroll
        for (int k = 0; k < 32; ++k) sTot += red[k];
        out[0] = sTot * (1.0f / 16.0f);  // each column sum replicated 16x
    }
}

extern "C" void kernel_launch(void* const* d_in, const int* in_sizes, int n_in,
                              void* d_out, int out_size, void* d_ws, size_t ws_size,
                              hipStream_t stream) {
    (void)in_sizes; (void)n_in; (void)out_size; (void)ws_size;
    const float* raw        = (const float*)d_in[0];
    const float* concat     = (const float*)d_in[1];
    const float* part       = (const float*)d_in[2];
    const float* top_n_prob = (const float*)d_in[3];
    const int*   targets    = (const int*)d_in[4];

    float* ws       = (float*)d_ws;
    float* ce_all   = ws;         // 2048 floats: [raw 256 | concat 256 | part 1536]
    float* part_nll = ws + 2048;  // 1536 floats
    float* out      = (float*)d_out;

    nts_row_stats<<<2048, BLK, 0, stream>>>(raw, concat, part, targets,
                                            ce_all, part_nll);
    nts_finalize<<<1, BLK, 0, stream>>>(part_nll, top_n_prob, ce_all, out);
}
